// TraditionalRNN_72619307041104
// MI455X (gfx1250) — matrix operand order repaired and checked by reading, not compiled, not run on hardware
//
#include <hip/hip_runtime.h>
#include <hip/hip_bf16.h>
#include <stdint.h>

// ---------------------------------------------------------------------------
// GRU on MI455X (gfx1250): bf16 WMMA (v_wmma_f32_16x16x32_bf16, fp32 accum).
// Phase 1: xi = inputs@Wi+bi as one parallel WMMA GEMM (bf16 out).
// Phase 2: persistent scan; 8 batch-tiles x 8 feature-chunks = 64 WGs.
//          h double-buffered in global as f32 (carry precision) + bf16 mirror;
//          bf16 mirror staged to LDS per step via GLOBAL_LOAD_ASYNC_TO_LDS_B128.
// Phase 3: head GEMM (async-staged carry) + carry copy.
// ---------------------------------------------------------------------------

typedef __attribute__((ext_vector_type(16))) __bf16 v16bf;
typedef __attribute__((ext_vector_type(8)))  float  v8f;

#define B_    128
#define T_    512
#define IN_   512
#define F_    1024
#define N3_   3072
#define OUT_  256

__device__ __forceinline__ unsigned short f2bf(float x) {
  unsigned u = __float_as_uint(x);
  u += 0x7FFFu + ((u >> 16) & 1u);          // round-to-nearest-even
  return (unsigned short)(u >> 16);
}
__device__ __forceinline__ unsigned pack2(float a, float b) {
  return (unsigned)f2bf(a) | ((unsigned)f2bf(b) << 16);
}
__device__ __forceinline__ float bf2f(unsigned short h) {
  return __uint_as_float(((unsigned)h) << 16);
}

union FragU { unsigned u[8]; uint4 q[2]; v16bf v; };

// Load one 32-byte fragment slice: two uint4 at p and p+4 (works for global
// and, after inlining/addrspace inference, for LDS -> ds_load_b128).
__device__ __forceinline__ v16bf load_frag(const unsigned* p) {
  FragU f;
  f.q[0] = *(const uint4*)(p);
  f.q[1] = *(const uint4*)(p + 4);
  return f.v;
}

__device__ __forceinline__ v8f wmma_bf16(v16bf a, v16bf b, v8f c) {
  return __builtin_amdgcn_wmma_f32_16x16x32_bf16(false, a, false, b,
                                                 (short)0, c, false, false);
}

// CDNA5 async copy: global -> LDS, 16B per lane, tracked by ASYNCcnt.
__device__ __forceinline__ void async_ld_b128(unsigned lds_off, const void* g) {
  asm volatile("global_load_async_to_lds_b128 %0, %1, off"
               :: "v"(lds_off), "v"(g) : "memory");
}
__device__ __forceinline__ void wait_async0() {
  asm volatile("s_wait_asynccnt 0x0" ::: "memory");
}

// ------------------------- preprocessing kernels ---------------------------

__global__ void k_cvt_pairs(const float* __restrict__ src,
                            unsigned* __restrict__ dst, int npairs) {
  int stride = gridDim.x * blockDim.x;
  for (int i = blockIdx.x * blockDim.x + threadIdx.x; i < npairs; i += stride)
    dst[i] = pack2(src[2 * i], src[2 * i + 1]);
}

// Pack row-major fp32 W[K][N] into bf16 B-fragment layout
// [ktile][ntile][lane][8 dwords]; lane=N%16 (+16 -> K half), VGPR v: K=2v,2v+1.
__global__ void k_pack_b(const float* __restrict__ W, unsigned* __restrict__ dst,
                         int K, int N) {
  int total  = (K * N) >> 1;
  int ntiles = N >> 4;
  int stride = gridDim.x * blockDim.x;
  for (int idx = blockIdx.x * blockDim.x + threadIdx.x; idx < total; idx += stride) {
    int v     = idx & 7;
    int lane  = (idx >> 3) & 31;
    int tile  = idx >> 8;
    int ntile = tile % ntiles;
    int ktile = tile / ntiles;
    int n = (ntile << 4) + (lane & 15);
    int k = (ktile << 5) + ((lane >> 4) << 4) + (v << 1);
    dst[idx] = pack2(W[(size_t)k * N + n], W[(size_t)(k + 1) * N + n]);
  }
}

__global__ void k_init(float* __restrict__ hA, unsigned* __restrict__ hbfA,
                       unsigned* __restrict__ ctr) {
  int stride = gridDim.x * blockDim.x;
  for (int i = blockIdx.x * blockDim.x + threadIdx.x; i < B_ * F_; i += stride)
    hA[i] = 0.0f;
  for (int i = blockIdx.x * blockDim.x + threadIdx.x; i < (B_ * F_) / 2; i += stride)
    hbfA[i] = 0u;
  if (blockIdx.x == 0 && threadIdx.x < 8) ctr[threadIdx.x] = 0u;
}

// ------------------- phase 1: xi = inputs @ Wi + bi (bf16 out) -------------
// M = B*T = 65536 (4096 m-tiles), N = 3072 (192 n-tiles), K = 512.
// One wave per (mtile, ntile-pair); 8 waves / 256-thread block.

__global__ void __launch_bounds__(256) k_gemm_xi(
    const unsigned* __restrict__ Xp,      // [65536][256] packed bf16 pairs
    const unsigned* __restrict__ WiP,     // frag-packed, ktiles=16 ntiles=192
    const float* __restrict__ bi,
    unsigned short* __restrict__ xi) {    // [65536][3072] bf16
  const int KSTR = 192 * 32 * 8;          // uints per ktile of packed Wi
  int w    = blockIdx.x * 8 + (threadIdx.x >> 5);
  int lane = threadIdx.x & 31;
  int mt   = w / 96;
  int ntp  = w - mt * 96;
  v8f acc0 = {}; v8f acc1 = {};
  // Per-lane bases; all k-loop addressing becomes immediate offsets.
  const unsigned* abase  = Xp + (size_t)(mt * 16 + (lane & 15)) * 256
                              + ((lane >> 4) << 3);
  const unsigned* b0base = WiP + (((size_t)(ntp * 2)     * 32 + lane) << 3);
  const unsigned* b1base = WiP + (((size_t)(ntp * 2 + 1) * 32 + lane) << 3);
#pragma unroll
  for (int kt = 0; kt < 16; ++kt) {
    v16bf a  = load_frag(abase  + kt * 16);
    v16bf b0 = load_frag(b0base + kt * KSTR);
    v16bf b1 = load_frag(b1base + kt * KSTR);
    acc0 = wmma_bf16(a, b0, acc0);
    acc1 = wmma_bf16(a, b1, acc1);
  }
  int n0 = ntp * 32 + (lane & 15);
  float bia0 = bi[n0], bia1 = bi[n0 + 16];
#pragma unroll
  for (int i = 0; i < 8; ++i) {
    int m = i + ((lane >> 4) << 3);
    size_t row = (size_t)mt * 16 + m;
    xi[row * N3_ + n0]      = f2bf(acc0[i] + bia0);
    xi[row * N3_ + n0 + 16] = f2bf(acc1[i] + bia1);
  }
}

// ----------------------- phase 2: persistent GRU scan ----------------------
// blockIdx: bt = >>3 (batch tile of 16 rows), fc = &7 (128-feature chunk).
// 128 threads = 4 waves; wave handles 2 feature n-tiles x 3 gates.

__global__ void __launch_bounds__(128) k_gru_scan(
    const unsigned short* __restrict__ xi,   // [B*T][3072] bf16
    const unsigned* __restrict__ WhP,        // frag-packed, ktiles=32 ntiles=192
    const float* __restrict__ bhn,           // [1024]
    float* __restrict__ hA, float* __restrict__ hB,
    unsigned short* __restrict__ hbfA,       // bf16 mirror of h, double-buffered
    unsigned short* __restrict__ hbfB,
    unsigned* __restrict__ ctr) {
  const int KSTR = 192 * 32 * 8;             // uints per ktile of packed Wh
  __shared__ __attribute__((aligned(128))) unsigned Hs[16 * 512]; // 16x1024 bf16
  int bt   = blockIdx.x >> 3;
  int fc   = blockIdx.x & 7;
  int wv   = threadIdx.x >> 5;
  int lane = threadIdx.x & 31;
  int fo   = fc << 7;
  int tid  = threadIdx.x;
  unsigned ldsb = (unsigned)(unsigned long long)(void*)Hs;

  // Per-lane invariant bases for A (LDS) and B (packed Wh) fragments.
  const unsigned* l_abase = &Hs[(lane & 15) * 512 + ((lane >> 4) << 3)];
  const unsigned* br0 = WhP + (((size_t)((fc << 3) + (wv << 1))       * 32 + lane) << 3);
  const unsigned* br1 = WhP + (((size_t)((fc << 3) + (wv << 1) + 1)   * 32 + lane) << 3);
  const int ZOFF = 64 * 32 * 8;              // +64 ntiles  (gate z)
  const int NOFF = 128 * 32 * 8;             // +128 ntiles (gate n)

  for (int t = 0; t < T_; ++t) {
    const float*          hcur   = (t & 1) ? hB : hA;
    float*                hnxt   = (t & 1) ? hA : hB;
    const unsigned short* hbfcur = (t & 1) ? hbfB : hbfA;
    unsigned short*       hbfnxt = (t & 1) ? hbfA : hbfB;

    // Async-stage this batch-tile's bf16 h block (32KB contiguous) into LDS.
    {
      const char* g = (const char*)(hbfcur + (size_t)bt * 16 * F_);
      for (int i = tid; i < 2048; i += 128)
        async_ld_b128(ldsb + (unsigned)i * 16u, g + (size_t)i * 16);
      wait_async0();
      __syncthreads();
    }

    // Prefetch next timestep's xi rows for this chunk while we compute.
    if (t + 1 < T_) {
      int m = tid & 15;
      size_t xrow = ((size_t)(bt * 16 + m) * T_ + (t + 1)) * N3_;
      __builtin_prefetch(&xi[xrow + fo], 0, 1);
      __builtin_prefetch(&xi[xrow + F_ + fo], 0, 1);
      __builtin_prefetch(&xi[xrow + 2 * F_ + fo], 0, 1);
    }

#pragma unroll
    for (int nt2 = 0; nt2 < 2; ++nt2) {
      int nt = (wv << 1) + nt2;                   // 0..7 inside chunk
      const unsigned* bb = nt2 ? br1 : br0;
      v8f ar = {}; v8f az = {}; v8f an = {};
#pragma unroll 8
      for (int kt = 0; kt < 32; ++kt) {
        v16bf a = load_frag(l_abase + kt * 16);   // a-frag reused x3
        ar = wmma_bf16(a, load_frag(bb + kt * KSTR),        ar);
        az = wmma_bf16(a, load_frag(bb + kt * KSTR + ZOFF), az);
        an = wmma_bf16(a, load_frag(bb + kt * KSTR + NOFF), an);
      }
      int n = lane & 15;
      int f = fo + (nt << 4) + n;
      float bh = bhn[f];
#pragma unroll
      for (int i = 0; i < 8; ++i) {
        int m = i + ((lane >> 4) << 3);
        int b = bt * 16 + m;
        size_t xrow = ((size_t)b * T_ + t) * N3_;
        float xr = bf2f(xi[xrow + f]);
        float xz = bf2f(xi[xrow + F_ + f]);
        float xn = bf2f(xi[xrow + 2 * F_ + f]);
        float hv = hcur[(size_t)b * F_ + f];
        float r  = 1.0f / (1.0f + __expf(-(xr + ar[i])));
        float z  = 1.0f / (1.0f + __expf(-(xz + az[i])));
        float nn = tanhf(xn + r * (an[i] + bh));
        float hn = (1.0f - z) * nn + z * hv;
        hnxt[(size_t)b * F_ + f]   = hn;          // f32 (carry precision)
        hbfnxt[(size_t)b * F_ + f] = f2bf(hn);    // bf16 mirror for next GEMM
      }
    }

    // 8-member barrier over this batch-tile's workgroups (monotonic counter).
    __syncthreads();
    __threadfence();
    if (tid == 0) {
      atomicAdd(&ctr[bt], 1u);
      unsigned target = 8u * (unsigned)(t + 1);
      while (__hip_atomic_load(&ctr[bt], __ATOMIC_RELAXED,
                               __HIP_MEMORY_SCOPE_AGENT) < target) {
        __builtin_amdgcn_s_sleep(1);
      }
    }
    __syncthreads();
    __threadfence();
  }
}

// ----------------------- phase 3: output head + carry ----------------------

__global__ void __launch_bounds__(128) k_head(
    const unsigned short* __restrict__ hbf,  // final carry, bf16 [128][1024]
    const unsigned* __restrict__ WoP,        // frag-packed, ktiles=32 ntiles=16
    const float* __restrict__ bo,
    float* __restrict__ out) {               // [128][256]
  const int KSTR = 16 * 32 * 8;              // uints per ktile of packed Wo
  __shared__ __attribute__((aligned(128))) unsigned Hs[16 * 512];
  int mt   = blockIdx.x;                     // 0..7
  int wv   = threadIdx.x >> 5;
  int lane = threadIdx.x & 31;
  unsigned ldsb = (unsigned)(unsigned long long)(void*)Hs;
  {
    const char* g = (const char*)(hbf + (size_t)mt * 16 * F_);
    for (int i = threadIdx.x; i < 2048; i += 128)
      async_ld_b128(ldsb + (unsigned)i * 16u, g + (size_t)i * 16);
    wait_async0();
    __syncthreads();
  }
  const unsigned* l_abase = &Hs[(lane & 15) * 512 + ((lane >> 4) << 3)];
  for (int nt4 = 0; nt4 < 4; ++nt4) {
    int nt = wv * 4 + nt4;
    const unsigned* bb = WoP + (((size_t)nt * 32 + lane) << 3);
    v8f acc = {};
#pragma unroll 8
    for (int kt = 0; kt < 32; ++kt) {
      v16bf a = load_frag(l_abase + kt * 16);
      acc = wmma_bf16(a, load_frag(bb + kt * KSTR), acc);
    }
    int n = nt * 16 + (lane & 15);
    float b = bo[n];
#pragma unroll
    for (int i = 0; i < 8; ++i) {
      int m = i + ((lane >> 4) << 3);
      out[(size_t)(mt * 16 + m) * OUT_ + n] = acc[i] + b;
    }
  }
}

__global__ void k_copy(const float* __restrict__ src, float* __restrict__ dst,
                       int n) {
  int stride = gridDim.x * blockDim.x;
  for (int i = blockIdx.x * blockDim.x + threadIdx.x; i < n; i += stride)
    dst[i] = src[i];
}

// ---------------------------------------------------------------------------

extern "C" void kernel_launch(void* const* d_in, const int* in_sizes, int n_in,
                              void* d_out, int out_size, void* d_ws, size_t ws_size,
                              hipStream_t stream) {
  (void)in_sizes; (void)n_in; (void)out_size; (void)ws_size;
  const float* inputs = (const float*)d_in[0];   // [128,512,512]
  const float* Wi     = (const float*)d_in[1];   // [512,3072]
  const float* bi     = (const float*)d_in[2];   // [3072]
  const float* Wh     = (const float*)d_in[3];   // [1024,3072]
  const float* bhn    = (const float*)d_in[4];   // [1024]
  const float* Wo     = (const float*)d_in[5];   // [1024,256]
  const float* bo     = (const float*)d_in[6];   // [256]
  float* out = (float*)d_out;                    // [128,256] ++ [128,1024]

  char* ws = (char*)d_ws;
  size_t off = 0;
  auto take = [&](size_t bytes) { char* p = ws + off;
                                  off += (bytes + 255) & ~(size_t)255; return p; };
  unsigned short* xi  = (unsigned short*)take((size_t)B_ * T_ * N3_ * 2);  // 402 MB
  unsigned* Xp  = (unsigned*)take((size_t)B_ * T_ * (IN_ / 2) * 4);        //  64 MB
  unsigned* WiP = (unsigned*)take((size_t)IN_ * N3_ * 2);
  unsigned* WhP = (unsigned*)take((size_t)F_ * N3_ * 2);
  unsigned* WoP = (unsigned*)take((size_t)F_ * OUT_ * 2);
  float* hA = (float*)take((size_t)B_ * F_ * 4);
  float* hB = (float*)take((size_t)B_ * F_ * 4);
  unsigned short* hbfA = (unsigned short*)take((size_t)B_ * F_ * 2);
  unsigned short* hbfB = (unsigned short*)take((size_t)B_ * F_ * 2);
  unsigned* ctr = (unsigned*)take(256);

  // Preprocess: bf16 conversion + fragment packing.
  k_cvt_pairs<<<4096, 256, 0, stream>>>(inputs, Xp, B_ * T_ * (IN_ / 2));
  k_pack_b<<<2048, 256, 0, stream>>>(Wi, WiP, IN_, N3_);
  k_pack_b<<<4096, 256, 0, stream>>>(Wh, WhP, F_, N3_);
  k_pack_b<<<512,  256, 0, stream>>>(Wo, WoP, F_, OUT_);
  k_init<<<512, 256, 0, stream>>>(hA, (unsigned*)hbfA, ctr);

  // Phase 1: 4096 m-tiles x 96 n-tile-pairs, 8 waves per block.
  k_gemm_xi<<<(4096 * 96) / 8, 256, 0, stream>>>(Xp, WiP, bi, xi);

  // Phase 2: persistent scan, 8 batch-tiles x 8 feature-chunks.
  k_gru_scan<<<64, 128, 0, stream>>>(xi, WhP, bhn, hA, hB, hbfA, hbfB, ctr);

  // Phase 3: head + carry (512 steps -> final h in hA / hbfA).
  k_head<<<8, 128, 0, stream>>>(hbfA, WoP, bo, out);
  k_copy<<<512, 256, 0, stream>>>(hA, out + B_ * OUT_, B_ * F_);
}
